// Net_1_2_3_21002390078204
// MI455X (gfx1250) — matrix operand order repaired and checked by reading, not compiled, not run on hardware
//
#include <hip/hip_runtime.h>
#include <hip/hip_bf16.h>
#include <cstdint>

// ---------------------------------------------------------------------------
// 1-2-3-GNN on MI455X (gfx1250, wave32, WMMA).
// Heavy math = bf16 WMMA (v_wmma_f32_16x16x32_bf16) with fp32 accumulation.
// NNConv edge-weight GEMM is fused: A-rows are outer(h_row, x_src_row) built
// on the fly from LDS (staged via global_load_async_to_lds_b128), avoiding
// ~2 GiB of We[E, mi*mo] HBM traffic. All B operands are pre-packed into the
// WMMA B-register layout so each lane fetches its fragment as 2x b128.
// ---------------------------------------------------------------------------

typedef __attribute__((ext_vector_type(16))) __bf16 v16bf;
typedef __attribute__((ext_vector_type(8)))  __bf16 v8bf;
typedef __attribute__((ext_vector_type(8)))  float  v8f;

__device__ __forceinline__ float eluf(float v) { return v > 0.f ? v : (__expf(v) - 1.f); }

__device__ __forceinline__ v16bf cat16(v8bf lo, v8bf hi) {
    return __builtin_shufflevector(lo, hi, 0, 1, 2, 3, 4, 5, 6, 7, 8, 9, 10, 11, 12, 13, 14, 15);
}

// ---------------- elementwise / utility kernels ----------------------------

__global__ void k_zero(float* p, int n) {
    int i = blockIdx.x * blockDim.x + threadIdx.x;
    if (i < n) p[i] = 0.f;
}

__global__ void k_cvt_bf16(const float* __restrict__ s, __bf16* __restrict__ d, int n) {
    int i = blockIdx.x * blockDim.x + threadIdx.x;
    if (i < n) d[i] = (__bf16)s[i];
}

// edge MLP layer 1: h = relu(ea @ W1 + b1), ea [E,7], W1 [7,128] -> bf16 [E,128]
__global__ __launch_bounds__(128) void k_edge_mlp(const float* __restrict__ ea,
                                                  const float* __restrict__ W1,
                                                  const float* __restrict__ b1,
                                                  __bf16* __restrict__ hE) {
    __shared__ float sea[7];
    int e = blockIdx.x, t = threadIdx.x;
    if (t < 7) sea[t] = ea[e * 7 + t];
    __syncthreads();
    float acc = b1[t];
#pragma unroll
    for (int j = 0; j < 7; ++j) acc += sea[j] * W1[j * 128 + t];
    hE[(size_t)e * 128 + t] = (__bf16)(acc > 0.f ? acc : 0.f);
}

// Pack W2 [128, MI*MO] (+ b2 rows, + zero pad) directly into the WMMA
// B-register layout: Bp[((kt*MOT + ct)*32 + lane)*16 + j] where the element j
// of lane holds K = kt*32 + (j<8 ? h*8+j : 16+h*8+j-8) (h = lane>>4), col = ct*16 + (lane&15), and the
// virtual K-row r maps r = k*MI+i -> W2[k, i*MO+o], rows [128MI,128MI+MI)=b2.
__global__ void k_pack_w2B(const float* __restrict__ W2, const float* __restrict__ b2,
                           __bf16* __restrict__ Bp, int MI, int MO, int Kp) {
    int MOT = MO >> 4;
    int total = (Kp >> 5) * MOT * 512;
    int i = blockIdx.x * blockDim.x + threadIdx.x;
    if (i >= total) return;
    int j = i & 15, lane = (i >> 4) & 31, tile = i >> 9;
    int kt = tile / MOT, ct = tile - kt * MOT;
    int r = kt * 32 + ((j < 8) ? ((lane >> 4) * 8 + j) : (16 + (lane >> 4) * 8 + j - 8));
    int o = ct * 16 + (lane & 15);
    int KV = 128 * MI;
    float v = 0.f;
    if (r < KV) {
        int k = r / MI, ii = r - k * MI;
        v = W2[(size_t)k * MI * MO + ii * MO + o];
    } else if (r < KV + MI) {
        v = b2[(r - KV) * MO + o];
    }
    Bp[i] = (__bf16)v;
}

// Pack fp32 weight W [K,N] (K mult of 32, N mult of 16) into WMMA B layout.
__global__ void k_pack_wB(const float* __restrict__ W, __bf16* __restrict__ Bp,
                          int K, int N) {
    int tilesN = N >> 4;
    int total = (K >> 5) * tilesN * 512;
    int i = blockIdx.x * blockDim.x + threadIdx.x;
    if (i >= total) return;
    int j = i & 15, lane = (i >> 4) & 31, tile = i >> 9;
    int kt = tile / tilesN, ct = tile - kt * tilesN;
    int r = kt * 32 + ((j < 8) ? ((lane >> 4) * 8 + j) : (16 + (lane >> 4) * 8 + j - 8));
    int col = ct * 16 + (lane & 15);
    Bp[i] = (__bf16)W[(size_t)r * N + col];
}

// ---------------- fused NNConv message kernel (WMMA) ------------------------
// msg[e,:] = (h[e,:] (x) x[src[e],:]) @ Wp   with bias rows folded into Wp.
// Each block: 64 edges (4 waves x 16 edges). h tile staged to LDS with the
// CDNA5 async-to-LDS data path; A tile synthesized from LDS rows.
template <int MI, int MO>
__global__ __launch_bounds__(128) void k_nnconv_msg(const __bf16* __restrict__ hE,
                                                    const __bf16* __restrict__ xbf,
                                                    const int* __restrict__ src,
                                                    const __bf16* __restrict__ Wp,
                                                    float* __restrict__ msg, int Kp) {
    constexpr int MI_SH = (MI == 16) ? 4 : ((MI == 32) ? 5 : 6);
    constexpr int KV = 128 * MI;
    constexpr int MOT = MO / 16;
    __shared__ __align__(16) __bf16 sh_h[64 * 128];
    __shared__ __align__(16) __bf16 sh_x[64 * MI];

    int tid = threadIdx.x;
    int eb = blockIdx.x * 64;

    // stage h rows: 64 edges x 128 bf16 = 16 KB = 1024 x b128 async chunks
#pragma unroll
    for (int it = 0; it < 8; ++it) {
        int c = it * 128 + tid;
        unsigned ldsoff = (unsigned)(uintptr_t)(sh_h + c * 8);
        const __bf16* gptr = hE + (size_t)eb * 128 + c * 8;
        asm volatile("global_load_async_to_lds_b128 %0, %1, off"
                     :
                     : "v"(ldsoff), "v"(gptr)
                     : "memory");
    }
    // stage gathered x rows (scattered; plain LDS writes)
    for (int i = tid; i < 64 * MI; i += 128) {
        int el = i / MI, ii = i - el * MI;
        sh_x[i] = xbf[(size_t)src[eb + el] * MI + ii];
    }
    asm volatile("s_wait_asynccnt 0x0" ::: "memory");
    __syncthreads();

    int wv = tid >> 5, lane = tid & 31;
    int m = lane & 15, half = lane >> 4;
    int eloc = wv * 16 + m;
    const int ka0 = half ? 8 : 0;    // A elems 0..7  -> K = ka0+j
    const int ka1 = half ? 24 : 16;  // A elems 8..15 -> K = ka1+(j-8)

    v8f acc[MOT];
#pragma unroll
    for (int t = 0; t < MOT; ++t)
#pragma unroll
        for (int j = 0; j < 8; ++j) acc[t][j] = 0.f;

    for (int kk = 0; kk < Kp; kk += 32) {
        v16bf a;
#pragma unroll
        for (int j = 0; j < 16; ++j) {
            int K = (j < 8) ? (ka0 + j) : (ka1 + j - 8);
            int r = kk + K;
            float av;
            if (r < KV)
                av = (float)sh_h[eloc * 128 + (r >> MI_SH)] *
                     (float)sh_x[eloc * MI + (r & (MI - 1))];
            else if (r < KV + MI)
                av = (float)sh_x[eloc * MI + (r - KV)];   // bias row contribution
            else
                av = 0.f;
            a[j] = (__bf16)av;
        }
        int kt = kk >> 5;
#pragma unroll
        for (int t = 0; t < MOT; ++t) {
            const __bf16* bp = Wp + (size_t)(((kt * MOT + t) * 32) + lane) * 16;
            v16bf b = cat16(*(const v8bf*)bp, *(const v8bf*)(bp + 8));
            acc[t] = __builtin_amdgcn_wmma_f32_16x16x32_bf16(
                false, a, false, b, (short)0, acc[t], false, false);
        }
    }
#pragma unroll
    for (int t = 0; t < MOT; ++t)
#pragma unroll
        for (int j = 0; j < 8; ++j) {
            int row = j + half * 8;
            msg[(size_t)(eb + wv * 16 + row) * MO + t * 16 + m] = acc[t][j];
        }
}

// ---------------- generic WMMA GEMM: C (?=/+=) A[M,K]bf16 @ Bp(packed) ------
// mode 0: C = A@W ; mode 2: C = elu(C + A@W + bias[col])
__global__ __launch_bounds__(256) void k_wmma_gemm(const __bf16* __restrict__ A,
                                                   const __bf16* __restrict__ Bp,
                                                   float* __restrict__ C, int M, int K,
                                                   int N, const float* __restrict__ bias,
                                                   int mode) {
    int tid = threadIdx.x;
    int lane = tid & 31, wv = tid >> 5;
    int tilesN = N >> 4;
    int tile = blockIdx.x * 8 + wv;
    int rt = tile / tilesN, ct = tile - rt * tilesN;
    if (rt * 16 >= M) return;  // wave-uniform: EXEC stays all-ones for WMMA
    int m = lane & 15, half = lane >> 4;
    const int ka0 = half ? 8 : 0, ka1 = half ? 24 : 16;

    v8f acc;
#pragma unroll
    for (int j = 0; j < 8; ++j) acc[j] = 0.f;

    const __bf16* Arow = A + (size_t)(rt * 16 + m) * K;
    for (int kk = 0; kk < K; kk += 32) {
        v16bf a = cat16(*(const v8bf*)(Arow + kk + ka0), *(const v8bf*)(Arow + kk + ka1));
        const __bf16* bp = Bp + (size_t)((((kk >> 5) * tilesN + ct) * 32) + lane) * 16;
        v16bf b = cat16(*(const v8bf*)bp, *(const v8bf*)(bp + 8));
        acc = __builtin_amdgcn_wmma_f32_16x16x32_bf16(false, a, false, b, (short)0, acc,
                                                      false, false);
    }
#pragma unroll
    for (int j = 0; j < 8; ++j) {
        int col = ct * 16 + m;
        size_t idx = (size_t)(rt * 16 + j + half * 8) * N + col;
        float v = acc[j];
        if (mode == 0) {
            C[idx] = v;
        } else {
            v += C[idx] + bias[col];
            C[idx] = eluf(v);
        }
    }
}

// ---------------- scatter / gather / pooling -------------------------------

__global__ void k_gather_scatter(const float* __restrict__ in, const int* __restrict__ sidx,
                                 const int* __restrict__ didx, float* out, int A, int F) {
    int i = blockIdx.x * blockDim.x + threadIdx.x;
    if (i >= A * F) return;
    int a = i / F, f = i - a * F;
    atomicAdd(&out[(size_t)didx[a] * F + f], in[(size_t)sidx[a] * F + f]);
}

__global__ void k_scatter_vals(const float* __restrict__ vals, const int* __restrict__ didx,
                               float* out, int E, int F) {
    int i = blockIdx.x * blockDim.x + threadIdx.x;
    if (i >= E * F) return;
    int e = i / F, f = i - e * F;
    atomicAdd(&out[(size_t)didx[e] * F + f], vals[i]);
}

__global__ void k_count(const int* __restrict__ idx, float* cnt, int A) {
    int i = blockIdx.x * blockDim.x + threadIdx.x;
    if (i < A) atomicAdd(&cnt[idx[i]], 1.f);
}

__global__ void k_pool_div_concat(const float* __restrict__ s, const float* __restrict__ cnt,
                                  const float* __restrict__ iso, float* __restrict__ h2,
                                  int Nc) {
    int i = blockIdx.x * blockDim.x + threadIdx.x;
    if (i >= Nc * 128) return;
    int n = i >> 7, j = i & 127;
    h2[i] = (j < 64) ? s[(size_t)n * 64 + j] / fmaxf(cnt[n], 1.f)
                     : iso[(size_t)n * 64 + (j - 64)];
}

__global__ void k_root_bias_elu(const float* __restrict__ x, const float* __restrict__ root,
                                const float* __restrict__ bias, const float* __restrict__ agg,
                                float* out, int Nn, int MI, int MO) {
    int i = blockIdx.x * blockDim.x + threadIdx.x;
    if (i >= Nn * MO) return;
    int n = i / MO, o = i - n * MO;
    float acc = agg[i] + bias[o];
    for (int k = 0; k < MI; ++k) acc += x[(size_t)n * MI + k] * root[k * MO + o];
    out[i] = eluf(acc);
}

__global__ void k_segment_sum(const float* __restrict__ in, const int* __restrict__ seg,
                              float* out, int Nn, int F) {
    int i = blockIdx.x * blockDim.x + threadIdx.x;
    if (i >= Nn * F) return;
    int n = i / F, f = i - n * F;
    atomicAdd(&out[(size_t)seg[n] * F + f], in[i]);
}

__global__ void k_concat6(const float* __restrict__ x1, const float* __restrict__ x2,
                          const float* __restrict__ x3, float* __restrict__ xc) {
    int i = blockIdx.x * blockDim.x + threadIdx.x;
    if (i >= 256 * 384) return;
    int b = i / 384, c = i - b * 384;
    int cc = c % 192, part = cc / 64, col = cc % 64;
    const float* s = (part == 0) ? x1 : ((part == 1) ? x2 : x3);
    xc[i] = s[b * 64 + col];
}

__global__ void k_fc(const float* __restrict__ in, const float* __restrict__ W,
                     const float* __restrict__ b, float* out, int M, int K, int N, int act) {
    int i = blockIdx.x * blockDim.x + threadIdx.x;
    if (i >= M * N) return;
    int m = i / N, n = i - m * N;
    float acc = b[n];
    for (int k = 0; k < K; ++k) acc += in[(size_t)m * K + k] * W[k * N + n];
    out[i] = act ? eluf(acc) : acc;
}

// ---------------------------------------------------------------------------

static inline int cdiv(int a, int b) { return (a + b - 1) / b; }

extern "C" void kernel_launch(void* const* d_in, const int* in_sizes, int n_in,
                              void* d_out, int out_size, void* d_ws, size_t ws_size,
                              hipStream_t stream) {
    const int N1 = 16384, E1 = 65536, NC = 65536;
    const int A2 = 131072, E2 = 262144, A3 = 196608, E3 = 262144;

    const float* x_in = (const float*)d_in[0];
    const int* ei1 = (const int*)d_in[1];
    const float* ea = (const float*)d_in[2];
    const int* batch = (const int*)d_in[3];
    const int* a2n = (const int*)d_in[4];
    const int* a2c = (const int*)d_in[5];
    const float* iso2 = (const float*)d_in[6];
    const int* ei2 = (const int*)d_in[7];
    const int* batch2 = (const int*)d_in[8];
    const int* a3n = (const int*)d_in[9];
    const int* a3c = (const int*)d_in[10];
    const float* iso3 = (const float*)d_in[11];
    const int* ei3 = (const int*)d_in[12];
    const int* batch3 = (const int*)d_in[13];
    const float* nnW1[3], *nnb1[3], *nnW2[3], *nnb2[3], *roots[3], *cbias[3];
    for (int c = 0; c < 3; ++c) {
        nnW1[c] = (const float*)d_in[14 + 6 * c + 0];
        nnb1[c] = (const float*)d_in[14 + 6 * c + 1];
        nnW2[c] = (const float*)d_in[14 + 6 * c + 2];
        nnb2[c] = (const float*)d_in[14 + 6 * c + 3];
        roots[c] = (const float*)d_in[14 + 6 * c + 4];
        cbias[c] = (const float*)d_in[14 + 6 * c + 5];
    }
    const float* Wrel4 = (const float*)d_in[32], *Wroot4 = (const float*)d_in[33], *b4 = (const float*)d_in[34];
    const float* Wrel5 = (const float*)d_in[35], *Wroot5 = (const float*)d_in[36], *b5 = (const float*)d_in[37];
    const float* Wrel6 = (const float*)d_in[38], *Wroot6 = (const float*)d_in[39], *b6 = (const float*)d_in[40];
    const float* Wrel7 = (const float*)d_in[41], *Wroot7 = (const float*)d_in[42], *b7 = (const float*)d_in[43];
    const float* fc1W = (const float*)d_in[44], *fc1b = (const float*)d_in[45];
    const float* fc2W = (const float*)d_in[46], *fc2b = (const float*)d_in[47];
    const float* fc3W = (const float*)d_in[48], *fc3b = (const float*)d_in[49];

    // ---- workspace bump allocator ----
    char* base = (char*)d_ws;
    size_t off = 0;
    auto alloc = [&](size_t bytes) -> void* {
        off = (off + 255) & ~(size_t)255;
        void* p = base + off;
        off += bytes;
        return p;
    };
    __bf16* hE   = (__bf16*)alloc((size_t)E1 * 128 * 2);
    __bf16* Wp   = (__bf16*)alloc((size_t)8256 * 64 * 2);
    float*  msg  = (float*)alloc((size_t)E1 * 64 * 4);
    float*  agg  = (float*)alloc((size_t)N1 * 64 * 4);
    float*  xa   = (float*)alloc((size_t)N1 * 64 * 4);
    float*  xb   = (float*)alloc((size_t)N1 * 64 * 4);
    __bf16* xbf  = (__bf16*)alloc((size_t)N1 * 64 * 2);
    float*  x1   = (float*)alloc(256 * 64 * 4);
    float*  x2   = (float*)alloc(256 * 64 * 4);
    float*  x3   = (float*)alloc(256 * 64 * 4);
    float*  s2   = (float*)alloc((size_t)NC * 64 * 4);
    float*  cnt  = (float*)alloc((size_t)NC * 4);
    float*  h2   = (float*)alloc((size_t)NC * 128 * 4);
    float*  aggG = (float*)alloc((size_t)NC * 128 * 4);
    __bf16* h2bf  = (__bf16*)alloc((size_t)NC * 128 * 2);
    __bf16* aggbf = (__bf16*)alloc((size_t)NC * 128 * 2);
    __bf16* Wlbf  = (__bf16*)alloc(128 * 64 * 2);
    __bf16* Wrbf  = (__bf16*)alloc(128 * 64 * 2);
    float*  gA   = (float*)alloc((size_t)NC * 64 * 4);
    float*  gB   = (float*)alloc((size_t)NC * 64 * 4);
    float*  xc   = (float*)alloc(256 * 384 * 4);
    float*  f1   = (float*)alloc(256 * 64 * 4);
    float*  f2   = (float*)alloc(256 * 32 * 4);

    const int* src1 = ei1;
    const int* dst1 = ei1 + E1;

    // ---- level 1: three NNConv layers ----
    const float* xcur = x_in;
    float* pp[2] = {xa, xb};
    for (int c = 0; c < 3; ++c) {
        int MI = (c == 0) ? 16 : ((c == 1) ? 32 : 64);
        int MO = (c == 0) ? 32 : 64;
        int Kp = ((128 * MI + MI + 31) / 32) * 32;

        k_edge_mlp<<<E1, 128, 0, stream>>>(ea, nnW1[c], nnb1[c], hE);
        int packn = (Kp >> 5) * (MO >> 4) * 512;
        k_pack_w2B<<<cdiv(packn, 256), 256, 0, stream>>>(nnW2[c], nnb2[c], Wp, MI, MO, Kp);
        k_cvt_bf16<<<cdiv(N1 * MI, 256), 256, 0, stream>>>(xcur, xbf, N1 * MI);
        if (MI == 16)
            k_nnconv_msg<16, 32><<<E1 / 64, 128, 0, stream>>>(hE, xbf, src1, Wp, msg, Kp);
        else if (MI == 32)
            k_nnconv_msg<32, 64><<<E1 / 64, 128, 0, stream>>>(hE, xbf, src1, Wp, msg, Kp);
        else
            k_nnconv_msg<64, 64><<<E1 / 64, 128, 0, stream>>>(hE, xbf, src1, Wp, msg, Kp);
        k_zero<<<cdiv(N1 * MO, 256), 256, 0, stream>>>(agg, N1 * MO);
        k_scatter_vals<<<cdiv(E1 * MO, 256), 256, 0, stream>>>(msg, dst1, agg, E1, MO);
        float* xnext = pp[c & 1];
        k_root_bias_elu<<<cdiv(N1 * MO, 256), 256, 0, stream>>>(xcur, roots[c], cbias[c],
                                                                agg, xnext, N1, MI, MO);
        xcur = xnext;
    }
    const float* h = xcur;  // [N1, 64]

    k_zero<<<cdiv(256 * 64, 256), 256, 0, stream>>>(x1, 256 * 64);
    k_segment_sum<<<cdiv(N1 * 64, 256), 256, 0, stream>>>(h, batch, x1, N1, 64);

    // ---- levels 2/3: avg-pool-by-assignment + two GraphConvs + segsum ----
    auto do_level = [&](const int* an, const int* ac, int A, const float* iso,
                        const int* ei, int Ecnt, const float* Wr1, const float* Wt1,
                        const float* gb1, const float* Wr2, const float* Wt2,
                        const float* gb2, const int* seg, float* xout) {
        k_zero<<<cdiv(NC * 64, 256), 256, 0, stream>>>(s2, NC * 64);
        k_zero<<<cdiv(NC, 256), 256, 0, stream>>>(cnt, NC);
        k_gather_scatter<<<cdiv(A * 64, 256), 256, 0, stream>>>(h, an, ac, s2, A, 64);
        k_count<<<cdiv(A, 256), 256, 0, stream>>>(ac, cnt, A);
        k_pool_div_concat<<<cdiv(NC * 128, 256), 256, 0, stream>>>(s2, cnt, iso, h2, NC);

        const int* s = ei;
        const int* d = ei + Ecnt;
        // GraphConv 1 (F=128 -> 64)
        k_zero<<<cdiv(NC * 128, 256), 256, 0, stream>>>(aggG, NC * 128);
        k_gather_scatter<<<cdiv(Ecnt * 128, 256), 256, 0, stream>>>(h2, s, d, aggG, Ecnt, 128);
        k_cvt_bf16<<<cdiv(NC * 128, 256), 256, 0, stream>>>(aggG, aggbf, NC * 128);
        k_cvt_bf16<<<cdiv(NC * 128, 256), 256, 0, stream>>>(h2, h2bf, NC * 128);
        k_pack_wB<<<cdiv(128 * 64, 256), 256, 0, stream>>>(Wr1, Wlbf, 128, 64);
        k_pack_wB<<<cdiv(128 * 64, 256), 256, 0, stream>>>(Wt1, Wrbf, 128, 64);
        int tiles = (NC / 16) * 4;
        k_wmma_gemm<<<tiles / 8, 256, 0, stream>>>(aggbf, Wlbf, gA, NC, 128, 64, nullptr, 0);
        k_wmma_gemm<<<tiles / 8, 256, 0, stream>>>(h2bf, Wrbf, gA, NC, 128, 64, gb1, 2);
        // GraphConv 2 (F=64 -> 64)
        k_zero<<<cdiv(NC * 64, 256), 256, 0, stream>>>(aggG, NC * 64);
        k_gather_scatter<<<cdiv(Ecnt * 64, 256), 256, 0, stream>>>(gA, s, d, aggG, Ecnt, 64);
        k_cvt_bf16<<<cdiv(NC * 64, 256), 256, 0, stream>>>(aggG, aggbf, NC * 64);
        k_cvt_bf16<<<cdiv(NC * 64, 256), 256, 0, stream>>>(gA, h2bf, NC * 64);
        k_pack_wB<<<cdiv(64 * 64, 256), 256, 0, stream>>>(Wr2, Wlbf, 64, 64);
        k_pack_wB<<<cdiv(64 * 64, 256), 256, 0, stream>>>(Wt2, Wrbf, 64, 64);
        k_wmma_gemm<<<tiles / 8, 256, 0, stream>>>(aggbf, Wlbf, gB, NC, 64, 64, nullptr, 0);
        k_wmma_gemm<<<tiles / 8, 256, 0, stream>>>(h2bf, Wrbf, gB, NC, 64, 64, gb2, 2);

        k_zero<<<cdiv(256 * 64, 256), 256, 0, stream>>>(xout, 256 * 64);
        k_segment_sum<<<cdiv(NC * 64, 256), 256, 0, stream>>>(gB, seg, xout, NC, 64);
    };
    do_level(a2n, a2c, A2, iso2, ei2, E2, Wrel4, Wroot4, b4, Wrel5, Wroot5, b5, batch2, x2);
    do_level(a3n, a3c, A3, iso3, ei3, E3, Wrel6, Wroot6, b6, Wrel7, Wroot7, b7, batch3, x3);

    // ---- head ----
    k_concat6<<<cdiv(256 * 384, 256), 256, 0, stream>>>(x1, x2, x3, xc);
    k_fc<<<cdiv(256 * 64, 256), 256, 0, stream>>>(xc, fc1W, fc1b, f1, 256, 384, 64, 1);
    k_fc<<<cdiv(256 * 32, 256), 256, 0, stream>>>(f1, fc2W, fc2b, f2, 256, 64, 32, 1);
    k_fc<<<1, 256, 0, stream>>>(f2, fc3W, fc3b, (float*)d_out, 256, 32, 1, 0);
}
